// STEventKGC_29162827940061
// MI455X (gfx1250) — matrix operand order repaired and checked by reading, not compiled, not run on hardware
//
#include <hip/hip_runtime.h>

// ============================================================================
// STEventKGC heterogeneous GNN for gfx1250 (MI455X).
// GEMMs: A pre-converted to f16 (one pass), B tile staged in LDS (f16,
// zero-padded); each wave computes 4 M-tiles sharing one B fragment with
// v_wmma_f32_16x16x32_f16. Edge gather/scatter: fp32 VMEM + float atomics.
// ============================================================================

typedef __attribute__((ext_vector_type(16))) _Float16 v16h;
typedef __attribute__((ext_vector_type(8)))  _Float16 v8h;
typedef __attribute__((ext_vector_type(8)))  float    v8f;

#define NE    100000
#define NS    50000
#define NG    2048
#define NSRC  1000
#define EOBS     1000000
#define ELOC     50000
#define EREVLOC  50000
#define EREVSRC  100000
#define EFROMSRC 100000
#define EPREV    300000
#define MAXK  128
#define MT    4        // M tiles per wave

// ---------------------------------------------------------------------------
// f32 -> f16 conversion pass (n must be a multiple of 8; true for all calls)
// ---------------------------------------------------------------------------
__global__ void k_cvt_f16(const float* __restrict__ x, _Float16* __restrict__ y, long n)
{
    const long i = ((long)blockIdx.x * blockDim.x + threadIdx.x) * 8;
    if (i >= n) return;
    const float4 a = *(const float4*)(x + i);
    const float4 b = *(const float4*)(x + i + 4);
    v8h o;
    o[0] = (_Float16)a.x; o[1] = (_Float16)a.y;
    o[2] = (_Float16)a.z; o[3] = (_Float16)a.w;
    o[4] = (_Float16)b.x; o[5] = (_Float16)b.y;
    o[6] = (_Float16)b.z; o[7] = (_Float16)b.w;
    *(v8h*)(y + i) = o;
}

// ---------------------------------------------------------------------------
// WMMA GEMM: Y[M,N] (+)= Xh[M,K](f16) @ W[N,K]^T (+ bias) (relu)
// Block = 256 threads = 8 waves; block owns one 16-wide N tile and 32
// consecutive M tiles (4 per wave, sharing the B fragment). The 16xK W slab
// is converted to f16 into LDS, column-major with K padded to a multiple of
// 32 (zero fill), so a B fragment is two 16B LDS reads.
//
// Fragment layouts (CDNA5 ISA 7.12.2, wave32):
//   A 16x32 f16 : M = lane&15 ; half j -> K = (j<8 ? j : j+8) + 8*(lane>=16)
//   B 32x16 f16 : N = lane&15 ; half j -> K = (j<8 ? j : j+8) + 8*(lane>=16)
//   C/D 16x16 f32: N = lane&15 ; vgpr r -> M = r + 8*(lane>=16)
//
// Call-site invariants: K % 16 == 0; buffers 16B-aligned.
// Rows >= M are address-clamped (their outputs are never stored).
// ---------------------------------------------------------------------------
__global__ void k_gemm_wmma(const _Float16* __restrict__ Xh,
                            const float* __restrict__ W,
                            const float* __restrict__ bias,
                            float* __restrict__ Y,
                            int M, int N, int K, int acc, int relu)
{
    __shared__ _Float16 ldsB[16 * MAXK];

    const int tid  = threadIdx.x;
    const int lane = tid & 31;
    const int wave = tid >> 5;
    const int tiles_n = (N + 15) >> 4;
    const int tiles_m = (M + 15) >> 4;
    const int tn  = blockIdx.x % tiles_n;
    const int tmg = blockIdx.x / tiles_n;
    const int Kp  = (K + 31) & ~31;            // padded K (zero-filled)

    // ---- cooperative stage of B tile into LDS (f16, column-major, padded) ----
    for (int e = tid; e < 16 * Kp; e += 256) {
        const int col = e / Kp;
        const int k   = e - col * Kp;
        const int wr  = tn * 16 + col;
        float v = 0.0f;
        if (wr < N && k < K) v = W[(long)wr * K + k];
        ldsB[col * Kp + k] = (_Float16)v;
    }
    __syncthreads();

    const int l15 = lane & 15;
    const int hi  = lane >> 4;
    const int tm0 = tmg * (8 * MT) + wave * MT;
    if (tm0 >= tiles_m) return;                // wave-uniform

    const _Float16* __restrict__ Bc = ldsB + l15 * Kp;
    const int cN = tn * 16 + l15;

    // per-tile A row pointers (clamped) and accumulators
    const _Float16* Ar[MT];
    v8f c[MT];
#pragma unroll
    for (int t = 0; t < MT; ++t) {
        int arow = (tm0 + t) * 16 + l15;
        if (arow > M - 1) arow = M - 1;
        Ar[t] = Xh + (long)arow * K;
#pragma unroll
        for (int r = 0; r < 8; ++r) c[t][r] = 0.0f;
    }
    if (acc) {
#pragma unroll
        for (int t = 0; t < MT; ++t) {
            const int cM0 = (tm0 + t) * 16 + hi * 8;
#pragma unroll
            for (int r = 0; r < 8; ++r) {
                const int m = cM0 + r;
                if (m < M && cN < N) c[t][r] = Y[(long)m * N + cN];
            }
        }
    }

    int k0 = 0;
    for (; k0 + 32 <= K; k0 += 32) {
        const v16h b = __builtin_shufflevector(*(const v8h*)(Bc + k0 + hi * 8), *(const v8h*)(Bc + k0 + hi * 8 + 16),
            0, 1, 2, 3, 4, 5, 6, 7, 8, 9, 10, 11, 12, 13, 14, 15);
#pragma unroll
        for (int t = 0; t < MT; ++t) {
            const v8h al = *(const v8h*)(Ar[t] + k0 + hi * 8);       // b128
            const v8h ah = *(const v8h*)(Ar[t] + k0 + hi * 8 + 16);  // b128
            const v16h a = __builtin_shufflevector(al, ah,
                0, 1, 2, 3, 4, 5, 6, 7, 8, 9, 10, 11, 12, 13, 14, 15);
            c[t] = __builtin_amdgcn_wmma_f32_16x16x32_f16(
                       false, a, false, b, (short)0, c[t], false, false);
        }
    }
    if (k0 < K) {                              // 16-wide K tail (K%32==16)
        const v16h b = __builtin_shufflevector(*(const v8h*)(Bc + k0 + hi * 8), *(const v8h*)(Bc + k0 + hi * 8 + 16),
            0, 1, 2, 3, 4, 5, 6, 7, 8, 9, 10, 11, 12, 13, 14, 15);
        const v8h zz = (v8h)(_Float16)0.0f;
#pragma unroll
        for (int t = 0; t < MT; ++t) {
            const v8h al = *(const v8h*)(Ar[t] + k0 + hi * 8);
            const v16h a = __builtin_shufflevector(al, zz,
                0, 1, 2, 3, 4, 5, 6, 7, 8, 9, 10, 11, 12, 13, 14, 15);
            c[t] = __builtin_amdgcn_wmma_f32_16x16x32_f16(
                       false, a, false, b, (short)0, c[t], false, false);
        }
    }

    const float bv = (bias != nullptr && cN < N) ? bias[cN] : 0.0f;
#pragma unroll
    for (int t = 0; t < MT; ++t) {
        const int cM0 = (tm0 + t) * 16 + hi * 8;
        if ((tm0 + t) * 16 + 16 <= M && tn * 16 + 16 <= N) {  // full tile
#pragma unroll
            for (int r = 0; r < 8; ++r) {
                float v = c[t][r] + bv;
                if (relu) v = fmaxf(v, 0.0f);
                Y[(long)(cM0 + r) * N + cN] = v;
            }
        } else {
#pragma unroll
            for (int r = 0; r < 8; ++r) {
                const int m = cM0 + r;
                if (m < M && cN < N) {
                    float v = c[t][r] + bv;
                    if (relu) v = fmaxf(v, 0.0f);
                    Y[(long)m * N + cN] = v;
                }
            }
        }
    }
}

// ---------------------------------------------------------------------------
// Elementwise / scatter helpers
// ---------------------------------------------------------------------------
__global__ void k_fill(float* __restrict__ p, float v, long n)
{
    const long i = (long)blockIdx.x * blockDim.x + threadIdx.x;
    if (i < n) p[i] = v;
}

__global__ void k_relu(float* __restrict__ p, long n)
{
    const long i = (long)blockIdx.x * blockDim.x + threadIdx.x;
    if (i < n) p[i] = fmaxf(p[i], 0.0f);
}

__global__ void k_add(float* __restrict__ d, const float* __restrict__ s, long n)
{
    const long i = (long)blockIdx.x * blockDim.x + threadIdx.x;
    if (i < n) d[i] += s[i];
}

// segment-sum of 128-wide source features + edge count per destination
__global__ void k_scatter_sum(const float* __restrict__ xsrc,
                              const int* __restrict__ es,
                              const int* __restrict__ ed,
                              int E,
                              float* __restrict__ msum,
                              float* __restrict__ cnt)
{
    const int e = blockIdx.x;
    if (e >= E) return;
    const int f = threadIdx.x;                 // 0..127
    const int s = es[e], d = ed[e];
    atomicAdd(&msum[(long)d * 128 + f], xsrc[(long)s * 128 + f]);
    if (f == 0) atomicAdd(&cnt[d], 1.0f);
}

__global__ void k_mean(float* __restrict__ msum, const float* __restrict__ cnt, long n128)
{
    const long i = (long)blockIdx.x * blockDim.x + threadIdx.x;
    if (i < n128) msum[i] /= fmaxf(cnt[i >> 7], 1.0f);
}

// time_mlp l1: out[e,t] = relu(W1[t]*attr[e] + b1[t]),  TE = 16
__global__ void k_time_l1(const float* __restrict__ attr,
                          const float* __restrict__ W1,
                          const float* __restrict__ b1,
                          float* __restrict__ out, int E)
{
    const long t = (long)blockIdx.x * blockDim.x + threadIdx.x;
    if (t >= (long)E * 16) return;
    const long e = t >> 4;
    const int  i = (int)(t & 15);
    out[t] = fmaxf(W1[i] * attr[e] + b1[i], 0.0f);
}

// order-preserving float <-> uint encoding for atomicMax over floats
__device__ __forceinline__ unsigned ord_enc(float f)
{
    const unsigned u = __float_as_uint(f);
    return (u & 0x80000000u) ? ~u : (u | 0x80000000u);
}
__device__ __forceinline__ float ord_dec(unsigned u)
{
    const unsigned b = (u & 0x80000000u) ? (u & 0x7fffffffu) : ~u;
    return __uint_as_float(b);
}

// alpha[e,h] = <q[d,h,:], k[s,h,:] + eproj[e,h,:]> / 8 ; segment-max via atomicMax
__global__ void k_alpha(const float* __restrict__ q,
                        const float* __restrict__ kf,
                        const float* __restrict__ eproj,
                        const int* __restrict__ es,
                        const int* __restrict__ ed,
                        int E,
                        float* __restrict__ alpha,
                        unsigned* __restrict__ amax)
{
    const long t = (long)blockIdx.x * blockDim.x + threadIdx.x;
    if (t >= 2L * E) return;
    const long e = t >> 1;
    const int  h = (int)(t & 1);
    const int  s = es[e], d = ed[e];
    const float4* qp = (const float4*)(q     + (long)d * 128 + h * 64);
    const float4* kp = (const float4*)(kf    + (long)s * 128 + h * 64);
    const float4* ep = (const float4*)(eproj + (long)e * 128 + h * 64);
    float a = 0.0f;
#pragma unroll 4
    for (int i = 0; i < 16; ++i) {
        const float4 qv = qp[i], kv = kp[i], ev = ep[i];
        a += qv.x * (kv.x + ev.x) + qv.y * (kv.y + ev.y)
           + qv.z * (kv.z + ev.z) + qv.w * (kv.w + ev.w);
    }
    a *= 0.125f;                               // 1/sqrt(64)
    alpha[t] = a;
    atomicMax(&amax[(long)d * 2 + h], ord_enc(a));
}

// exp(alpha - max) in place + segment-sum denominator
__global__ void k_exden(float* __restrict__ alpha,
                        const unsigned* __restrict__ amax,
                        const int* __restrict__ ed,
                        int E,
                        float* __restrict__ den)
{
    const long t = (long)blockIdx.x * blockDim.x + threadIdx.x;
    if (t >= 2L * E) return;
    const long e = t >> 1;
    const int  h = (int)(t & 1);
    const int  d = ed[e];
    const float m  = ord_dec(amax[(long)d * 2 + h]);
    const float ex = expf(alpha[t] - m);
    alpha[t] = ex;
    atomicAdd(&den[(long)d * 2 + h], ex);
}

// out[d,f] += w[e,h] * (v[s,f] + eproj[e,f]),  h = f/64
__global__ void k_attn_scatter(const float* __restrict__ vf,
                               const float* __restrict__ eproj,
                               const float* __restrict__ alpha,
                               const float* __restrict__ den,
                               const int* __restrict__ es,
                               const int* __restrict__ ed,
                               int E,
                               float* __restrict__ out)
{
    const int e = blockIdx.x;
    if (e >= E) return;
    const int f = threadIdx.x;                 // 0..127
    const int h = f >> 6;
    const int s = es[e], d = ed[e];
    const float w = alpha[(long)e * 2 + h] / fmaxf(den[(long)d * 2 + h], 1e-16f);
    atomicAdd(&out[(long)d * 128 + f],
              w * (vf[(long)s * 128 + f] + eproj[(long)e * 128 + f]));
}

// ---------------------------------------------------------------------------
// Host-side orchestration
// ---------------------------------------------------------------------------
extern "C" void kernel_launch(void* const* d_in, const int* in_sizes, int n_in,
                              void* d_out, int out_size, void* d_ws, size_t ws_size,
                              hipStream_t stream)
{
    (void)in_sizes; (void)out_size; (void)ws_size;

    // ---- positional inputs (setup_inputs insertion order) ----
    const float* x_event  = (const float*)d_in[0];
    const float* x_sensor = (const float*)d_in[1];
    const float* x_geo    = (const float*)d_in[2];
    const float* x_src    = (const float*)d_in[3];
    const int* ei_obs     = (const int*)d_in[4];
    const int* ei_loc     = (const int*)d_in[5];
    const int* ei_revloc  = (const int*)d_in[6];
    const int* ei_revsrc  = (const int*)d_in[7];
    const int* ei_fromsrc = (const int*)d_in[8];
    const int* ei_prev    = (const int*)d_in[9];
    const float* attr     = (const float*)d_in[10];

    // ---- params pytree flattened JAX-style (dicts sorted, lists in order) ----
    // top level: band_head, geo_head, layers[0..1], node_lin, time_mlp
    // heads: l1.W,l1.b,l2.W,l2.b ; layer rel order:
    //   fromsrc(l.W,l.b,r.W), loc(..), obs(..),
    //   prev(e.W,k.W,k.b,q.W,q.b,skip.W,skip.b,v.W,v.b),
    //   revloc(..), revsrc(..)
    const float* P[96];
    for (int i = 0; i < n_in && i < 96; ++i) P[i] = (const float*)d_in[i];
    const int BAND = 11, GEO = 15;          // head param bases
    const int LBASE[2] = {19, 43};          // per-layer base, stride 24
    const int NL = 67, TM = 75;             // node_lin, time_mlp bases

    // ---- workspace carving ----
    float* w = (float*)d_ws;
    size_t o = 0;
    auto carve = [&](size_t n) { float* p = w + o; o += n; return p; };
    float* xe   = carve((size_t)NE * 128);
    float* xs   = carve((size_t)NS * 128);
    float* xg   = carve((size_t)NG * 128);
    float* xr   = carve((size_t)NSRC * 128);
    float* ne_  = carve((size_t)NE * 128);
    float* ns_  = carve((size_t)NS * 128);
    float* ng_  = carve((size_t)NG * 128);
    float* nr_  = carve((size_t)NSRC * 128);
    float* eenc1 = carve((size_t)EPREV * 16);
    float* eenc  = carve((size_t)EPREV * 16);
    float* eproj = carve((size_t)EPREV * 128);
    float* qb   = carve((size_t)NE * 128);
    float* kb   = carve((size_t)NE * 128);
    float* vb   = carve((size_t)NE * 128);
    float* msum = carve((size_t)NE * 128);
    float* cnt  = carve((size_t)NE);
    float* alph = carve((size_t)EPREV * 2);
    float* amaxf= carve((size_t)NE * 2);
    float* den  = carve((size_t)NE * 2);
    float* attn = carve((size_t)NE * 128);
    float* hid  = carve((size_t)NE * 128);
    // f16 scratch for GEMM A operand: max M*K = max(NE*128, EPREV*16) halves
    _Float16* xh = (_Float16*)carve((size_t)NE * 128 / 2 + 8);
    unsigned* amax = (unsigned*)amaxf;

    float* out_geo  = (float*)d_out;
    float* out_band = out_geo + (size_t)NE * 2048;

    // ---- launch helpers ----
    auto gemm = [&](const float* X, const float* W_, const float* B, float* Y,
                    int M, int N, int K, int acc, int relu) {
        const long nA = (long)M * K;
        k_cvt_f16<<<(unsigned)((nA / 8 + 255) / 256), 256, 0, stream>>>(X, xh, nA);
        const long tiles_m = (M + 15) / 16;
        const long tiles_n = (N + 15) / 16;
        const unsigned grid =
            (unsigned)(tiles_n * ((tiles_m + 8 * MT - 1) / (8 * MT)));
        k_gemm_wmma<<<grid, 256, 0, stream>>>(xh, W_, B, Y, M, N, K, acc, relu);
    };
    auto fill = [&](float* p, float v, long n) {
        k_fill<<<(unsigned)((n + 255) / 256), 256, 0, stream>>>(p, v, n);
    };

    // ================= node_lin projections =================
    gemm(x_event,  P[NL + 0], P[NL + 1], xe, NE,   128, 64, 0, 0);
    gemm(x_geo,    P[NL + 2], P[NL + 3], xg, NG,   128, 16, 0, 0);
    gemm(x_sensor, P[NL + 4], P[NL + 5], xs, NS,   128, 32, 0, 0);
    gemm(x_src,    P[NL + 6], P[NL + 7], xr, NSRC, 128, 16, 0, 0);

    // ================= time MLP =================
    k_time_l1<<<(unsigned)(((long)EPREV * 16 + 255) / 256), 256, 0, stream>>>(
        attr, P[TM + 0], P[TM + 1], eenc1, EPREV);
    gemm(eenc1, P[TM + 2], P[TM + 3], eenc, EPREV, 16, 16, 0, 0);

    float *ce = xe, *cs = xs, *cg = xg, *cr = xr;
    float *De = ne_, *Ds = ns_, *Dg = ng_, *Dr = nr_;

    for (int l = 0; l < 2; ++l) {
        const int B = LBASE[l];
        const int FS = B + 0, LOC = B + 3, OBS = B + 6, PRV = B + 9,
                  RL = B + 18, RS = B + 21;

        // ---- event: SAGE(obs: sensor -> event) ----
        fill(msum, 0.0f, (long)NE * 128); fill(cnt, 0.0f, NE);
        k_scatter_sum<<<EOBS, 128, 0, stream>>>(cs, ei_obs, ei_obs + EOBS, EOBS, msum, cnt);
        k_mean<<<(unsigned)(((long)NE * 128 + 255) / 256), 256, 0, stream>>>(msum, cnt, (long)NE * 128);
        gemm(msum, P[OBS + 0], P[OBS + 1], De, NE, 128, 128, 0, 0);   // obs.l
        gemm(ce,   P[OBS + 2], nullptr,    De, NE, 128, 128, 1, 0);   // obs.r

        // ---- event: SAGE(revsrc: source -> event) ----
        fill(msum, 0.0f, (long)NE * 128); fill(cnt, 0.0f, NE);
        k_scatter_sum<<<EREVSRC, 128, 0, stream>>>(cr, ei_revsrc, ei_revsrc + EREVSRC, EREVSRC, msum, cnt);
        k_mean<<<(unsigned)(((long)NE * 128 + 255) / 256), 256, 0, stream>>>(msum, cnt, (long)NE * 128);
        gemm(msum, P[RS + 0], P[RS + 1], De, NE, 128, 128, 1, 0);
        gemm(ce,   P[RS + 2], nullptr,   De, NE, 128, 128, 1, 0);

        // ---- event: TransformerConv(prev) ----
        gemm(ce,   P[PRV + 3], P[PRV + 4], qb,    NE,    128, 128, 0, 0);  // q
        gemm(ce,   P[PRV + 1], P[PRV + 2], kb,    NE,    128, 128, 0, 0);  // k
        gemm(ce,   P[PRV + 7], P[PRV + 8], vb,    NE,    128, 128, 0, 0);  // v
        gemm(eenc, P[PRV + 0], nullptr,    eproj, EPREV, 128, 16,  0, 0);  // e
        fill(amaxf, 0.0f, (long)NE * 2);     // 0u == lowest ordered encoding
        fill(den,   0.0f, (long)NE * 2);
        k_alpha<<<(unsigned)((2L * EPREV + 255) / 256), 256, 0, stream>>>(
            qb, kb, eproj, ei_prev, ei_prev + EPREV, EPREV, alph, amax);
        k_exden<<<(unsigned)((2L * EPREV + 255) / 256), 256, 0, stream>>>(
            alph, amax, ei_prev + EPREV, EPREV, den);
        fill(attn, 0.0f, (long)NE * 128);
        k_attn_scatter<<<EPREV, 128, 0, stream>>>(
            vb, eproj, alph, den, ei_prev, ei_prev + EPREV, EPREV, attn);
        k_add<<<(unsigned)(((long)NE * 128 + 255) / 256), 256, 0, stream>>>(De, attn, (long)NE * 128);
        gemm(ce, P[PRV + 5], P[PRV + 6], De, NE, 128, 128, 1, 0);          // skip

        // ---- geocell: SAGE(loc: sensor -> geocell) ----
        fill(msum, 0.0f, (long)NG * 128); fill(cnt, 0.0f, NG);
        k_scatter_sum<<<ELOC, 128, 0, stream>>>(cs, ei_loc, ei_loc + ELOC, ELOC, msum, cnt);
        k_mean<<<(unsigned)(((long)NG * 128 + 255) / 256), 256, 0, stream>>>(msum, cnt, (long)NG * 128);
        gemm(msum, P[LOC + 0], P[LOC + 1], Dg, NG, 128, 128, 0, 0);
        gemm(cg,   P[LOC + 2], nullptr,    Dg, NG, 128, 128, 1, 0);

        // ---- sensor: SAGE(revloc: geocell -> sensor) ----
        fill(msum, 0.0f, (long)NS * 128); fill(cnt, 0.0f, NS);
        k_scatter_sum<<<EREVLOC, 128, 0, stream>>>(cg, ei_revloc, ei_revloc + EREVLOC, EREVLOC, msum, cnt);
        k_mean<<<(unsigned)(((long)NS * 128 + 255) / 256), 256, 0, stream>>>(msum, cnt, (long)NS * 128);
        gemm(msum, P[RL + 0], P[RL + 1], Ds, NS, 128, 128, 0, 0);
        gemm(cs,   P[RL + 2], nullptr,   Ds, NS, 128, 128, 1, 0);

        // ---- source: SAGE(fromsrc: event -> source) ----
        fill(msum, 0.0f, (long)NSRC * 128); fill(cnt, 0.0f, NSRC);
        k_scatter_sum<<<EFROMSRC, 128, 0, stream>>>(ce, ei_fromsrc, ei_fromsrc + EFROMSRC, EFROMSRC, msum, cnt);
        k_mean<<<(unsigned)(((long)NSRC * 128 + 255) / 256), 256, 0, stream>>>(msum, cnt, (long)NSRC * 128);
        gemm(msum, P[FS + 0], P[FS + 1], Dr, NSRC, 128, 128, 0, 0);
        gemm(cr,   P[FS + 2], nullptr,   Dr, NSRC, 128, 128, 1, 0);

        // ---- relu + swap ----
        k_relu<<<(unsigned)(((long)NE * 128 + 255) / 256), 256, 0, stream>>>(De, (long)NE * 128);
        k_relu<<<(unsigned)(((long)NS * 128 + 255) / 256), 256, 0, stream>>>(Ds, (long)NS * 128);
        k_relu<<<(unsigned)(((long)NG * 128 + 255) / 256), 256, 0, stream>>>(Dg, (long)NG * 128);
        k_relu<<<(unsigned)(((long)NSRC * 128 + 255) / 256), 256, 0, stream>>>(Dr, (long)NSRC * 128);
        float* t;
        t = ce; ce = De; De = t;
        t = cs; cs = Ds; Ds = t;
        t = cg; cg = Dg; Dg = t;
        t = cr; cr = Dr; Dr = t;
    }

    // ================= output heads =================
    gemm(ce,  P[GEO + 0],  P[GEO + 1],  hid,      NE, 128,  128, 0, 1);  // geo l1 + relu
    gemm(hid, P[GEO + 2],  P[GEO + 3],  out_geo,  NE, 2048, 128, 0, 0);  // geo l2
    gemm(ce,  P[BAND + 0], P[BAND + 1], hid,      NE, 128,  128, 0, 1);  // band l1 + relu
    gemm(hid, P[BAND + 2], P[BAND + 3], out_band, NE, 16,   128, 0, 0);  // band l2
}
